// TransformerDecoderLayer_5712306503706
// MI455X (gfx1250) — compile-verified, matrix operand order checked by reading, not run on hardware
//
#include <hip/hip_runtime.h>
#include <hip/hip_bf16.h>
#include <cmath>

// B=8, Qn=1024, Kn=2048, QD=512, KVD=256, FFN=2048, H=8
typedef _Float16 v16h __attribute__((ext_vector_type(16)));
typedef _Float16 v8h  __attribute__((ext_vector_type(8)));
typedef float    v8f  __attribute__((ext_vector_type(8)));

#define WMMA_F32_F16(a, b, c) \
  __builtin_amdgcn_wmma_f32_16x16x32_f16(false, (a), false, (b), (short)0, (c), false, false)

__device__ __forceinline__ v16h cat8(v8h lo, v8h hi) {
  return __builtin_shufflevector(lo, hi, 0,1,2,3,4,5,6,7,8,9,10,11,12,13,14,15);
}

// Issue one per-lane 16B async copy global->LDS (ASYNCcnt-tracked).
__device__ __forceinline__ void async_b128(const _Float16* g, const _Float16* lds) {
  unsigned ldsaddr = (unsigned)(unsigned long long)(const void*)lds;  // low 32b = LDS offset
  unsigned long long gaddr = (unsigned long long)(const void*)g;
  asm volatile("global_load_async_to_lds_b128 %0, %1, off" :: "v"(ldsaddr), "v"(gaddr) : "memory");
}

// ---------------------------------------------------------------------------
// Weight pre-pass: Wt[n*K + k] = (f16) W[k*N + n]  (32x32 LDS-tiled transpose)
// ---------------------------------------------------------------------------
__global__ __launch_bounds__(256) void wt_transpose_kernel(
    const float* __restrict__ W, _Float16* __restrict__ Wt, int K, int N) {
  __shared__ float tile[32][33];
  const int n0 = blockIdx.x * 32, k0 = blockIdx.y * 32;
  const int tx = threadIdx.x, ty = threadIdx.y;
#pragma unroll
  for (int j = 0; j < 4; ++j)
    tile[ty + j * 8][tx] = W[(size_t)(k0 + ty + j * 8) * N + n0 + tx];
  __syncthreads();
#pragma unroll
  for (int j = 0; j < 4; ++j)
    Wt[(size_t)(n0 + ty + j * 8) * K + k0 + tx] = (_Float16)tile[tx][ty + j * 8];
}

// ---------------------------------------------------------------------------
// GEMM: C[M,N] = act(A[M,K] @ W[K,N] + bias[N]).
// W is pre-transposed f16 (Wt[N,K]) and staged to LDS with async-to-LDS b128.
// A is f32 (cvt staging) or f16 (async staging), templated. Output f32 or f16.
// Tile 128x128x32, double-buffered LDS, 8 waves (2x4), 64x32 per wave.
// ---------------------------------------------------------------------------
template <typename TA>
__device__ __forceinline__ void stage_tile(const TA* A, const _Float16* Wt,
                                           _Float16* sA, _Float16* sB,
                                           int m0, int n0, int K, int k0, int t) {
  // W tile: 128 n-rows x 32 k (64B/row) = 512 x 16B chunks, 2 per thread, async
#pragma unroll
  for (int hlf = 0; hlf < 2; ++hlf) {
    int c = t + hlf * 256;
    int n = c >> 2, q = c & 3;
    async_b128(Wt + (size_t)(n0 + n) * K + k0 + q * 8, &sB[n * 40 + q * 8]);
  }
  if constexpr (sizeof(TA) == 2) {
    // f16 A: identical async staging
#pragma unroll
    for (int hlf = 0; hlf < 2; ++hlf) {
      int c = t + hlf * 256;
      int r = c >> 2, q = c & 3;
      async_b128((const _Float16*)A + (size_t)(m0 + r) * K + k0 + q * 8,
                 &sA[r * 40 + q * 8]);
    }
  } else {
    // f32 A: load + packed cvt + ds_store
    int r = t >> 1, hh = t & 1;
    const float* asrc = (const float*)A + (size_t)(m0 + r) * K + k0 + hh * 16;
    _Float16* adst = &sA[r * 40 + hh * 16];
#pragma unroll
    for (int i = 0; i < 16; ++i) adst[i] = (_Float16)asrc[i];
    __builtin_prefetch(asrc + 32, 0, 3);
  }
}

template <typename TA, bool RELU, typename TO>
__global__ __launch_bounds__(256) void gemm_kernel(
    const TA* __restrict__ A, const _Float16* __restrict__ Wt,
    const float* __restrict__ bias, TO* __restrict__ C,
    int M, int N, int K) {
  __shared__ __align__(16) _Float16 sA[2][128 * 40];  // 40-half pitch: 16B-aligned rows
  __shared__ __align__(16) _Float16 sB[2][128 * 40];

  const int t    = threadIdx.x;
  const int m0   = blockIdx.y * 128;
  const int n0   = blockIdx.x * 128;
  const int wid  = t >> 5, lane = t & 31;
  const int wm   = wid >> 2, wn = wid & 3;
  const int ln   = lane & 15, hf = lane >> 4;

  v8f acc[4][2] = {};
  const int nk = K >> 5;

  stage_tile<TA>(A, Wt, sA[0], sB[0], m0, n0, K, 0, t);
  for (int kt = 0; kt < nk; ++kt) {
    asm volatile("s_wait_asynccnt 0x0" ::: "memory");  // tile kt fully in LDS
    __syncthreads();
    if (kt + 1 < nk)
      stage_tile<TA>(A, Wt, sA[(kt + 1) & 1], sB[(kt + 1) & 1], m0, n0, K,
                     (kt + 1) * 32, t);
    const _Float16* cA = sA[kt & 1];
    const _Float16* cB = sB[kt & 1];

    v16h aop[4];
#pragma unroll
    for (int s = 0; s < 4; ++s) {
      int r = wm * 64 + s * 16 + ln;
      aop[s] = cat8(*(const v8h*)&cA[r * 40 + hf * 8],
                    *(const v8h*)&cA[r * 40 + hf * 8 + 16]);
    }
#pragma unroll
    for (int u = 0; u < 2; ++u) {
      int cl = wn * 32 + u * 16 + ln;
      v16h bop = cat8(*(const v8h*)&cB[cl * 40 + hf * 8],
                      *(const v8h*)&cB[cl * 40 + hf * 8 + 16]);
#pragma unroll
      for (int s = 0; s < 4; ++s) acc[s][u] = WMMA_F32_F16(aop[s], bop, acc[s][u]);
    }
    __syncthreads();  // all reads of buf[kt&1] done before it is restaged
  }

#pragma unroll
  for (int u = 0; u < 2; ++u) {
    int col = n0 + wn * 32 + u * 16 + ln;
    float bv = bias[col];
#pragma unroll
    for (int s = 0; s < 4; ++s) {
#pragma unroll
      for (int v = 0; v < 8; ++v) {
        int row = m0 + wm * 64 + s * 16 + v + 8 * hf;
        float val = acc[s][u][v] + bv;
        if (RELU) val = fmaxf(val, 0.0f);
        C[(size_t)row * N + col] = (TO)val;
      }
    }
  }
}

// ---------------------------------------------------------------------------
// Flash attention over f16 q/k/v, one wave per 16-row q tile, 32 keys/iter.
// DH=64 add-mode (SA: q=qa+qb), DH=128 concat-mode (CA: per-head [qa|qb]).
// ---------------------------------------------------------------------------
template <int DH, bool CONCAT>
__global__ __launch_bounds__(128) void flash_attn_kernel(
    const _Float16* __restrict__ qa, int qa_str, const _Float16* __restrict__ qb, int qb_str,
    const _Float16* __restrict__ ka, int ka_str, const _Float16* __restrict__ kb, int kb_str,
    const _Float16* __restrict__ vv, int v_str,
    const unsigned char* __restrict__ mask,
    float* __restrict__ out, int out_str,
    int qlen, int klen, float scale) {
  constexpr int NC = DH / 32;
  __shared__ __align__(16) _Float16 sP[4][16 * 40];

  const int wid  = threadIdx.x >> 5, lane = threadIdx.x & 31;
  const int ln   = lane & 15, hf = lane >> 4;
  const int h    = blockIdx.y, b = blockIdx.z;
  const int q0   = (blockIdx.x * 4 + wid) * 16;
  const size_t qbase = (size_t)b * qlen;
  const size_t kbase = (size_t)b * klen;

  // Q operands: vectorized v8h loads (16B aligned by construction)
  v16h qop[NC];
  {
    const _Float16* ra = qa + (qbase + q0 + ln) * (size_t)qa_str + h * 64;
    const _Float16* rb = qb + (qbase + q0 + ln) * (size_t)qb_str + h * 64;
#pragma unroll
    for (int c = 0; c < NC; ++c) {
      if (CONCAT) {
        const _Float16* p = (c < NC / 2) ? ra + c * 32 : rb + (c - NC / 2) * 32;
        qop[c] = cat8(*(const v8h*)(p + hf * 8), *(const v8h*)(p + hf * 8 + 16));
      } else {
        v8h lo = *(const v8h*)(ra + c * 32 + hf * 8) + *(const v8h*)(rb + c * 32 + hf * 8);
        v8h hi = *(const v8h*)(ra + c * 32 + hf * 8 + 16) + *(const v8h*)(rb + c * 32 + hf * 8 + 16);
        qop[c] = cat8(lo, hi);
      }
    }
  }

  float mst[8], lst[8];
#pragma unroll
  for (int v = 0; v < 8; ++v) { mst[v] = -1e30f; lst[v] = 0.0f; }
  v8f acc[4] = {};

  for (int kk = 0; kk < klen; kk += 32) {
    v8f sS[2] = {};
    float madd[2];
#pragma unroll
    for (int j = 0; j < 2; ++j) {
      const size_t krow = kbase + kk + j * 16 + ln;  // K row == B-operand column
      const _Float16* ra = ka + krow * (size_t)ka_str + h * 64;
      const _Float16* rb = kb + krow * (size_t)kb_str + h * 64;
#pragma unroll
      for (int c = 0; c < NC; ++c) {
        v16h kop;
        if (CONCAT) {
          const _Float16* p = (c < NC / 2) ? ra + c * 32 : rb + (c - NC / 2) * 32;
          kop = cat8(*(const v8h*)(p + hf * 8), *(const v8h*)(p + hf * 8 + 16));
        } else {
          v8h lo = *(const v8h*)(ra + c * 32 + hf * 8) + *(const v8h*)(rb + c * 32 + hf * 8);
          v8h hi = *(const v8h*)(ra + c * 32 + hf * 8 + 16) + *(const v8h*)(rb + c * 32 + hf * 8 + 16);
          kop = cat8(lo, hi);
        }
        sS[j] = WMMA_F32_F16(qop[c], kop, sS[j]);
      }
      madd[j] = mask[(size_t)b * klen + kk + j * 16 + ln] ? -1e30f : 0.0f;
    }

    // Online softmax; row stats reduce over the 16-lane half holding that row
#pragma unroll
    for (int v = 0; v < 8; ++v) {
      float s0 = sS[0][v] * scale + madd[0];
      float s1 = sS[1][v] * scale + madd[1];
      float mx = fmaxf(s0, s1);
#pragma unroll
      for (int m = 1; m <= 8; m <<= 1) mx = fmaxf(mx, __shfl_xor(mx, m, 32));
      float mn = fmaxf(mst[v], mx);
      float al = __expf(mst[v] - mn);
      mst[v] = mn;
      float p0 = __expf(s0 - mn);
      float p1 = __expf(s1 - mn);
      float rs = p0 + p1;
#pragma unroll
      for (int m = 1; m <= 8; m <<= 1) rs += __shfl_xor(rs, m, 32);
      lst[v] = lst[v] * al + rs;
#pragma unroll
      for (int u = 0; u < 4; ++u) acc[u][v] *= al;
      int Mr = v + 8 * hf;
      sP[wid][Mr * 40 + ln]      = (_Float16)p0;
      sP[wid][Mr * 40 + 16 + ln] = (_Float16)p1;
    }
    asm volatile("s_wait_dscnt 0x0" ::: "memory");  // wave-local LDS RAW fence

    v16h pop = cat8(*(const v8h*)&sP[wid][ln * 40 + hf * 8],
                    *(const v8h*)&sP[wid][ln * 40 + hf * 8 + 16]);
#pragma unroll
    for (int u = 0; u < 4; ++u) {
      int vcol = h * 64 + u * 16 + ln;
      v16h vop;
#pragma unroll
      for (int i = 0; i < 16; ++i) {
        int kr = kk + hf * 8 + (i & 7) + ((i >= 8) ? 16 : 0);
        vop[i] = vv[(kbase + kr) * (size_t)v_str + vcol];
      }
      acc[u] = WMMA_F32_F16(pop, vop, acc[u]);
    }
  }

#pragma unroll
  for (int u = 0; u < 4; ++u) {
    int col = h * 64 + u * 16 + ln;
#pragma unroll
    for (int v = 0; v < 8; ++v) {
      int row = q0 + v + 8 * hf;
      out[(qbase + row) * (size_t)out_str + col] = acc[u][v] / fmaxf(lst[v], 1e-30f);
    }
  }
}

// ---------------------------------------------------------------------------
// Fused residual-add + LayerNorm over D=512; optional f16 shadow output.
// ---------------------------------------------------------------------------
__device__ __forceinline__ float block_sum_256(float v, float* red) {
#pragma unroll
  for (int m = 16; m >= 1; m >>= 1) v += __shfl_xor(v, m, 32);
  int wid = threadIdx.x >> 5, lane = threadIdx.x & 31;
  if (lane == 0) red[wid] = v;
  __syncthreads();
  if (wid == 0) {
    float s = (lane < 8) ? red[lane] : 0.0f;
#pragma unroll
    for (int m = 4; m >= 1; m >>= 1) s += __shfl_xor(s, m, 32);
    if (lane == 0) red[0] = s;
  }
  __syncthreads();
  float r = red[0];
  __syncthreads();
  return r;
}

__global__ __launch_bounds__(256) void add_ln_kernel(
    const float* __restrict__ x, const float* __restrict__ r,
    const float* __restrict__ g, const float* __restrict__ bta,
    float* __restrict__ out, _Float16* __restrict__ out16) {
  __shared__ float red[8];
  const int t = threadIdx.x;
  const size_t base = (size_t)blockIdx.x * 512;
  float a0 = x[base + t]       + r[base + t];
  float a1 = x[base + 256 + t] + r[base + 256 + t];
  float mean = block_sum_256(a0 + a1, red) * (1.0f / 512.0f);
  float d0 = a0 - mean, d1 = a1 - mean;
  float var = block_sum_256(d0 * d0 + d1 * d1, red) * (1.0f / 512.0f);
  float inv = rsqrtf(var + 1e-5f);
  float o0 = d0 * inv * g[t]       + bta[t];
  float o1 = d1 * inv * g[t + 256] + bta[t + 256];
  out[base + t]       = o0;
  out[base + 256 + t] = o1;
  if (out16 != nullptr) {
    out16[base + t]       = (_Float16)o0;
    out16[base + 256 + t] = (_Float16)o1;
  }
}

// ---------------------------------------------------------------------------
// Orchestration
// ---------------------------------------------------------------------------
extern "C" void kernel_launch(void* const* d_in, const int* in_sizes, int n_in,
                              void* d_out, int out_size, void* d_ws, size_t ws_size,
                              hipStream_t stream) {
  constexpr int Bc = 8, Qn = 1024, Kn = 2048, Hc = 8;
  constexpr size_t MQ = 8192, MK = 16384;

  const float* queries    = (const float*)d_in[0];
  const unsigned char* qm = (const unsigned char*)d_in[1];
  const float* box_pos    = (const float*)d_in[2];
  const float* center_pos = (const float*)d_in[3];
  const float* kv_data    = (const float*)d_in[4];
  const unsigned char* km = (const unsigned char*)d_in[5];
  const float* kv_pos     = (const float*)d_in[6];
  const float* sa_qkv_w   = (const float*)d_in[7];
  const float* sa_qkv_b   = (const float*)d_in[8];
  const float* sa_pos_w   = (const float*)d_in[9];
  const float* sa_pos_b   = (const float*)d_in[10];
  const float* sa_ln_g    = (const float*)d_in[11];
  const float* sa_ln_b    = (const float*)d_in[12];
  const float* ca_q_w     = (const float*)d_in[13];
  const float* ca_q_b     = (const float*)d_in[14];
  const float* ca_kv_w    = (const float*)d_in[15];
  const float* ca_kv_b    = (const float*)d_in[16];
  const float* ca_qpos_w  = (const float*)d_in[17];
  const float* ca_qpos_b  = (const float*)d_in[18];
  const float* ca_kpos_w  = (const float*)d_in[19];
  const float* ca_kpos_b  = (const float*)d_in[20];
  const float* ca_ln_g    = (const float*)d_in[21];
  const float* ca_ln_b    = (const float*)d_in[22];
  const float* ffn_w1     = (const float*)d_in[23];
  const float* ffn_b1     = (const float*)d_in[24];
  const float* ffn_w2     = (const float*)d_in[25];
  const float* ffn_b2     = (const float*)d_in[26];
  const float* ffn_ln_g   = (const float*)d_in[27];
  const float* ffn_ln_b   = (const float*)d_in[28];

  // --- workspace byte layout with liveness-based reuse (~152 MB) ---
  constexpr size_t OFF_QKV16   = 0;                                   // f16 MQ*1536
  constexpr size_t OFF_POS16   = OFF_QKV16 + MQ * 1536 * 2;           // f16 MQ*1024
  constexpr size_t OFF_ATTN    = OFF_POS16 + MQ * 1024 * 2;           // f32 MQ*512
  constexpr size_t END_R0      = OFF_ATTN + MQ * 512 * 4;
  constexpr size_t OFF_SAOUT   = END_R0;                              // f32 (long-lived)
  constexpr size_t OFF_SAOUT16 = OFF_SAOUT + MQ * 512 * 4;            // f16 (long-lived)
  constexpr size_t END_SA      = OFF_SAOUT16 + MQ * 512 * 2;
  constexpr size_t OFF_CAQ16   = 0;                                   // reuse region 0
  constexpr size_t OFF_CAQP16  = OFF_CAQ16 + MQ * 512 * 2;
  constexpr size_t OFF_CAKV16  = OFF_CAQP16 + MQ * 512 * 2;           // ends < END_R0
  constexpr size_t OFF_CAKP16  = END_SA;
  constexpr size_t OFF_CAATTN  = OFF_CAKP16 + MK * 512 * 2;           // f32
  constexpr size_t OFF_CAOUT   = OFF_CAATTN + MQ * 512 * 4;           // f32 (long-lived)
  constexpr size_t OFF_CAOUT16 = OFF_CAOUT + MQ * 512 * 4;            // f16 (long-lived)
  constexpr size_t END_CA      = OFF_CAOUT16 + MQ * 512 * 2;
  constexpr size_t OFF_FFN1_16 = 0;                                   // reuse region 0
  constexpr size_t OFF_FFN2    = OFF_FFN1_16 + MQ * 2048 * 2;         // f32
  constexpr size_t OFF_WT_SAQKV = END_CA;                             // f16 Wt[N,K] pool
  constexpr size_t OFF_WT_SAPOS = OFF_WT_SAQKV + (size_t)1536 * 512 * 2;
  constexpr size_t OFF_WT_CAQ   = OFF_WT_SAPOS + (size_t)1024 * 1024 * 2;
  constexpr size_t OFF_WT_CAKV  = OFF_WT_CAQ + (size_t)512 * 512 * 2;
  constexpr size_t OFF_WT_CAQP  = OFF_WT_CAKV + (size_t)1024 * 256 * 2;
  constexpr size_t OFF_WT_CAKP  = OFF_WT_CAQP + (size_t)512 * 512 * 2;
  constexpr size_t OFF_WT_FFN1  = OFF_WT_CAKP + (size_t)512 * 256 * 2;
  constexpr size_t OFF_WT_FFN2  = OFF_WT_FFN1 + (size_t)2048 * 512 * 2;

  char* wsb = (char*)d_ws;
  auto WSF = [&](size_t off) { return (float*)(wsb + off); };
  auto WSH = [&](size_t off) { return (_Float16*)(wsb + off); };
  float* outp = (float*)d_out;

  dim3 blk(256);
  dim3 tblk(32, 8);

  // ---- weight pre-pass: f32 [K,N] -> f16 transposed [N,K] ----
  wt_transpose_kernel<<<dim3(1536 / 32, 512 / 32),  tblk, 0, stream>>>(sa_qkv_w,  WSH(OFF_WT_SAQKV), 512, 1536);
  wt_transpose_kernel<<<dim3(1024 / 32, 1024 / 32), tblk, 0, stream>>>(sa_pos_w,  WSH(OFF_WT_SAPOS), 1024, 1024);
  wt_transpose_kernel<<<dim3(512 / 32, 512 / 32),   tblk, 0, stream>>>(ca_q_w,    WSH(OFF_WT_CAQ),   512, 512);
  wt_transpose_kernel<<<dim3(1024 / 32, 256 / 32),  tblk, 0, stream>>>(ca_kv_w,   WSH(OFF_WT_CAKV),  256, 1024);
  wt_transpose_kernel<<<dim3(512 / 32, 512 / 32),   tblk, 0, stream>>>(ca_qpos_w, WSH(OFF_WT_CAQP),  512, 512);
  wt_transpose_kernel<<<dim3(512 / 32, 256 / 32),   tblk, 0, stream>>>(ca_kpos_w, WSH(OFF_WT_CAKP),  256, 512);
  wt_transpose_kernel<<<dim3(2048 / 32, 512 / 32),  tblk, 0, stream>>>(ffn_w1,    WSH(OFF_WT_FFN1),  512, 2048);
  wt_transpose_kernel<<<dim3(512 / 32, 2048 / 32),  tblk, 0, stream>>>(ffn_w2,    WSH(OFF_WT_FFN2),  2048, 512);

  // ---- self-attention ----
  gemm_kernel<float, false, _Float16><<<dim3(1536 / 128, MQ / 128), blk, 0, stream>>>(
      queries, WSH(OFF_WT_SAQKV), sa_qkv_b, WSH(OFF_QKV16), (int)MQ, 1536, 512);
  gemm_kernel<float, false, _Float16><<<dim3(1024 / 128, MQ / 128), blk, 0, stream>>>(
      box_pos, WSH(OFF_WT_SAPOS), sa_pos_b, WSH(OFF_POS16), (int)MQ, 1024, 1024);
  flash_attn_kernel<64, false><<<dim3(Qn / 64, Hc, Bc), 128, 0, stream>>>(
      WSH(OFF_QKV16), 1536, WSH(OFF_POS16), 1024,              // q = qkv[:, :512] + pos[:, :512]
      WSH(OFF_QKV16) + 512, 1536, WSH(OFF_POS16) + 512, 1024,  // k
      WSH(OFF_QKV16) + 1024, 1536,                             // v = qkv[:, 1024:]
      qm, WSF(OFF_ATTN), 512, Qn, Qn, 0.125f);                 // 1/sqrt(64)
  add_ln_kernel<<<MQ, blk, 0, stream>>>(queries, WSF(OFF_ATTN), sa_ln_g, sa_ln_b,
                                        WSF(OFF_SAOUT), WSH(OFF_SAOUT16));

  // ---- cross-attention projections ----
  gemm_kernel<_Float16, false, _Float16><<<dim3(512 / 128, MQ / 128), blk, 0, stream>>>(
      WSH(OFF_SAOUT16), WSH(OFF_WT_CAQ), ca_q_b, WSH(OFF_CAQ16), (int)MQ, 512, 512);
  gemm_kernel<float, false, _Float16><<<dim3(512 / 128, MQ / 128), blk, 0, stream>>>(
      center_pos, WSH(OFF_WT_CAQP), ca_qpos_b, WSH(OFF_CAQP16), (int)MQ, 512, 512);
  gemm_kernel<float, false, _Float16><<<dim3(1024 / 128, MK / 128), blk, 0, stream>>>(
      kv_data, WSH(OFF_WT_CAKV), ca_kv_b, WSH(OFF_CAKV16), (int)MK, 1024, 256);
  gemm_kernel<float, false, _Float16><<<dim3(512 / 128, MK / 128), blk, 0, stream>>>(
      kv_pos, WSH(OFF_WT_CAKP), ca_kpos_b, WSH(OFF_CAKP16), (int)MK, 512, 256);
  // ---- cross-attention (per-head concat of content + pos, dh=128) ----
  flash_attn_kernel<128, true><<<dim3(Qn / 64, Hc, Bc), 128, 0, stream>>>(
      WSH(OFF_CAQ16), 512, WSH(OFF_CAQP16), 512,
      WSH(OFF_CAKV16), 1024, WSH(OFF_CAKP16), 512,
      WSH(OFF_CAKV16) + 512, 1024,                             // v = ca_kv[:, 512:]
      km, WSF(OFF_CAATTN), 512, Qn, Kn, 0.0883883476483184f);  // 1/sqrt(128)
  add_ln_kernel<<<MQ, blk, 0, stream>>>(WSF(OFF_SAOUT), WSF(OFF_CAATTN), ca_ln_g,
                                        ca_ln_b, WSF(OFF_CAOUT), WSH(OFF_CAOUT16));

  // ---- FFN ----
  gemm_kernel<_Float16, true, _Float16><<<dim3(2048 / 128, MQ / 128), blk, 0, stream>>>(
      WSH(OFF_CAOUT16), WSH(OFF_WT_FFN1), ffn_b1, WSH(OFF_FFN1_16), (int)MQ, 2048, 512);
  gemm_kernel<_Float16, false, float><<<dim3(512 / 128, MQ / 128), blk, 0, stream>>>(
      WSH(OFF_FFN1_16), WSH(OFF_WT_FFN2), ffn_b2, WSF(OFF_FFN2), (int)MQ, 512, 2048);
  add_ln_kernel<<<MQ, blk, 0, stream>>>(WSF(OFF_CAOUT), WSF(OFF_FFN2), ffn_ln_g,
                                        ffn_ln_b, outp, (_Float16*)nullptr);

  (void)in_sizes; (void)n_in; (void)out_size; (void)ws_size;
}
